// MultiHeadAttention_29265907155096
// MI455X (gfx1250) — matrix operand order repaired and checked by reading, not compiled, not run on hardware
//
#include <hip/hip_runtime.h>

typedef __bf16 bf16;
typedef __attribute__((ext_vector_type(16))) __bf16 v16bf;
typedef __attribute__((ext_vector_type(8)))  __bf16 v8bf;
typedef __attribute__((ext_vector_type(4)))  __bf16 v4bf;
typedef __attribute__((ext_vector_type(8)))  float  v8f;
typedef __attribute__((ext_vector_type(4)))  int    v4i;

union AFrag { v16bf v; v8bf h[2]; };

static __device__ __forceinline__ v8f wmma_bf16(v16bf a, v16bf b, v8f c) {
  // D = A(16x32 bf16) * B(32x16 bf16) + C(16x16 f32)
  return __builtin_amdgcn_wmma_f32_16x16x32_bf16(false, a, false, b, (short)0, c,
                                                 false, false);
}

// ---------------------------------------------------------------------------
// Async global->LDS staging (GLOBAL_LOAD_ASYNC_TO_LDS_B128, ASYNCcnt-tracked).
// Builtin signature (from hipcc diagnostic): (v4i AS1*, v4i AS3*, imm, imm).
// Guarded so a missing builtin falls back to plain LDS stores.
// ---------------------------------------------------------------------------
#if defined(__has_builtin)
#if __has_builtin(__builtin_amdgcn_global_load_async_to_lds_b128)
#define HAVE_ASYNC_LDS 1
#endif
#endif
#ifndef HAVE_ASYNC_LDS
#define HAVE_ASYNC_LDS 0
#endif

typedef __attribute__((address_space(1))) v4i g_v4i;
typedef __attribute__((address_space(3))) v4i l_v4i;

static __device__ __forceinline__ void copy16_to_lds(const bf16* g, bf16* l) {
#if HAVE_ASYNC_LDS
  __builtin_amdgcn_global_load_async_to_lds_b128((g_v4i*)g, (l_v4i*)l,
                                                 /*offset=*/0, /*cpol=*/0);
#else
  *reinterpret_cast<v8bf*>(l) = *reinterpret_cast<const v8bf*>(g);
#endif
}

static __device__ __forceinline__ void async_wait_all() {
#if HAVE_ASYNC_LDS
#if __has_builtin(__builtin_amdgcn_s_wait_asynccnt)
  __builtin_amdgcn_s_wait_asynccnt(0);
#else
  asm volatile("s_wait_asynccnt 0x0" ::: "memory");
#endif
#endif
}

// ---------------------------------------------------------------------------
// f32 -> bf16 cast (all element counts are multiples of 4)
// ---------------------------------------------------------------------------
__global__ void cast_kernel(const float* __restrict__ in, bf16* __restrict__ out,
                            long n) {
  long i = ((long)blockIdx.x * blockDim.x + threadIdx.x) * 4;
  if (i + 3 < n) {
    const float4 f = *reinterpret_cast<const float4*>(in + i);
    v4bf o;
    o[0] = (bf16)f.x; o[1] = (bf16)f.y; o[2] = (bf16)f.z; o[3] = (bf16)f.w;
    *reinterpret_cast<v4bf*>(out + i) = o;
  }
}

// ---------------------------------------------------------------------------
// C[M,N] = A[M,K] * B[N,K]^T   (both row-major, K contiguous -> NT GEMM)
// Block tile 128x64, BK=32, 256 threads = 8 waves (4 along M x 2 along N),
// each wave owns a 32x32 tile = 2x2 WMMA accumulators.
// Double-buffered LDS; tiles staged with async global->LDS copies.
// ---------------------------------------------------------------------------
template <bool OUT_F32>
__global__ __launch_bounds__(256) void gemm_nt(const bf16* __restrict__ A,
                                               const bf16* __restrict__ Bm,
                                               void* __restrict__ Cout,
                                               int N, int K) {
  constexpr int LDS_A = 72;  // halves; 144B row stride, 16B aligned
  constexpr int LDS_B = 72;
  __shared__ __align__(16) bf16 As[2][128 * LDS_A];
  __shared__ __align__(16) bf16 Bs[2][64 * LDS_B];

  const int tid  = threadIdx.x;
  const int lane = tid & 31;
  const int wid  = tid >> 5;
  const int l16  = lane & 15;
  const int lh   = lane >> 4;

  const int bm = blockIdx.y * 128;
  const int bn = blockIdx.x * 64;
  const int wm = (wid >> 1) * 32;
  const int wn = (wid & 1) * 32;

  v8f acc[2][2];
#pragma unroll
  for (int mi = 0; mi < 2; ++mi)
#pragma unroll
    for (int ni = 0; ni < 2; ++ni)
#pragma unroll
      for (int j = 0; j < 8; ++j) acc[mi][ni][j] = 0.0f;

  // global->LDS mapping (no bounds checks: every dim divides the tiles)
  const int arow = tid >> 1, acol = (tid & 1) * 16;  // 128 rows x 32 halves
  const int brow = tid >> 2, bcol = (tid & 3) * 8;   //  64 rows x 32 halves
  const bf16* aptr = A  + (size_t)(bm + arow) * K + acol;
  const bf16* bptr = Bm + (size_t)(bn + brow) * K + bcol;

  auto stage = [&](int buf, int kb) {
    copy16_to_lds(aptr + kb,     &As[buf][arow * LDS_A + acol]);
    copy16_to_lds(aptr + kb + 8, &As[buf][arow * LDS_A + acol + 8]);
    copy16_to_lds(bptr + kb,     &Bs[buf][brow * LDS_B + bcol]);
  };

  stage(0, 0);
  int ib = 0;
  for (int kb = 0; kb < K; kb += 32, ib ^= 1) {
    async_wait_all();    // my async copies into As[ib]/Bs[ib] are done
    __syncthreads();     // everyone's copies done; prev reads of other buf done
    if (kb + 32 < K) stage(ib ^ 1, kb + 32);

    AFrag af[2], bfr[2];
#pragma unroll
    for (int mi = 0; mi < 2; ++mi) {
      // A 16x32 bf16 layout: lane holds row l16; K halves {lh*8..+7, lh*8+16..+23}
      const bf16* p = &As[ib][(wm + mi * 16 + l16) * LDS_A + lh * 8];
      af[mi].h[0] = *reinterpret_cast<const v8bf*>(p);
      af[mi].h[1] = *reinterpret_cast<const v8bf*>(p + 16);
    }
#pragma unroll
    for (int ni = 0; ni < 2; ++ni) {
      // B 32x16 bf16 layout: lane holds column l16; K halves {lh*8..+7, lh*8+16..+23}
      const bf16* p = &Bs[ib][(wn + ni * 16 + l16) * LDS_B + lh * 8];
      bfr[ni].h[0] = *reinterpret_cast<const v8bf*>(p);
      bfr[ni].h[1] = *reinterpret_cast<const v8bf*>(p + 16);
    }
#pragma unroll
    for (int mi = 0; mi < 2; ++mi)
#pragma unroll
      for (int ni = 0; ni < 2; ++ni)
        acc[mi][ni] = wmma_bf16(af[mi].v, bfr[ni].v, acc[mi][ni]);
  }

  // C layout: lane -> col bn+wn+ni*16+l16 ; VGPR r -> row bm+wm+mi*16+lh*8+r
#pragma unroll
  for (int mi = 0; mi < 2; ++mi)
#pragma unroll
    for (int ni = 0; ni < 2; ++ni)
#pragma unroll
      for (int r = 0; r < 8; ++r) {
        const size_t row = (size_t)(bm + wm + mi * 16 + lh * 8 + r);
        const size_t col = (size_t)(bn + wn + ni * 16 + l16);
        if constexpr (OUT_F32)
          ((float*)Cout)[row * N + col] = acc[mi][ni][r];
        else
          ((bf16*)Cout)[row * N + col] = (bf16)acc[mi][ni][r];
      }
}

// ---------------------------------------------------------------------------
// Flash attention: one block (4 waves, 128 thr) per (b, h, 64-query block).
// qkv rows: [B*S, 6144] where cols [0,2048)=Q, [2048,4096)=K, [4096,6144)=V,
// head h occupies 128 contiguous cols inside each third.
// ---------------------------------------------------------------------------
__global__ __launch_bounds__(128) void flash_attn(const bf16* __restrict__ qkv,
                                                  bf16* __restrict__ attout) {
  constexpr int S = 2048, TD = 6144;
  constexpr int KS_LD = 136;  // [key][dk]  64 x 128(+8)
  constexpr int VT_LD = 72;   // [dk][key] 128 x  64(+8)
  constexpr int PS_LD = 72;   // per-wave P staging 16 x 64(+8)
  __shared__ __align__(16) bf16 Ks[64 * KS_LD];
  __shared__ __align__(16) bf16 Vt[128 * VT_LD];
  __shared__ __align__(16) bf16 Ps[4 * 16 * PS_LD];

  const int tid  = threadIdx.x;
  const int lane = tid & 31;
  const int wid  = tid >> 5;
  const int l16  = lane & 15;
  const int lh   = lane >> 4;

  const int qblk = blockIdx.x;  // 0..31
  const int h    = blockIdx.y;  // 0..15
  const int b    = blockIdx.z;  // 0..1

  // Q fragments (16 rows per wave, DK=128 -> 4 k-steps), straight from global
  AFrag qf[4];
  {
    const size_t qrow = (size_t)(b * S + qblk * 64 + wid * 16 + l16);
    const bf16* qbase = qkv + qrow * TD + h * 128;
#pragma unroll
    for (int ks = 0; ks < 4; ++ks) {
      const int off = ks * 32 + lh * 8;
      qf[ks].h[0] = *reinterpret_cast<const v8bf*>(qbase + off);
      qf[ks].h[1] = *reinterpret_cast<const v8bf*>(qbase + off + 16);
    }
  }

  float m[8], ssum[8];
  v8f o[8];
#pragma unroll
  for (int r = 0; r < 8; ++r) { m[r] = -1e30f; ssum[r] = 0.0f; }
#pragma unroll
  for (int n = 0; n < 8; ++n)
#pragma unroll
    for (int j = 0; j < 8; ++j) o[n][j] = 0.0f;

  const float scale = 0.08838834764831845f;  // 1/sqrt(128)

  const int krow = tid >> 1;
  const int kcol = (tid & 1) * 64;

  for (int kb2 = 0; kb2 <= qblk; ++kb2) {  // causal: skip blocks above diag
    __syncthreads();  // previous iteration's reads of Ks/Vt are done
    {  // stage K [key][dk] (async) and V transposed [dk][key]
      const size_t rbase = (size_t)(b * S + kb2 * 64 + krow) * TD + h * 128;
      const bf16* ksrc = qkv + rbase + 2048 + kcol;
#pragma unroll
      for (int i = 0; i < 8; ++i)
        copy16_to_lds(ksrc + i * 8, &Ks[krow * KS_LD + kcol + i * 8]);
      const bf16* vsrc = qkv + rbase + 4096 + kcol;
#pragma unroll
      for (int i = 0; i < 8; ++i) {
        v8bf t = *reinterpret_cast<const v8bf*>(vsrc + i * 8);
#pragma unroll
        for (int j = 0; j < 8; ++j)
          Vt[(kcol + i * 8 + j) * VT_LD + krow] = t[j];
      }
      if (kb2 < qblk) {  // prefetch next causal block (64 rows ahead)
        __builtin_prefetch(ksrc + (size_t)64 * TD, 0, 0);
        __builtin_prefetch(vsrc + (size_t)64 * TD, 0, 0);
      }
    }
    async_wait_all();
    __syncthreads();

    // scores S = Q K^T * scale : 4 n-tiles (64 keys) x 4 k-steps
    float sc[4][8];
    const bool diag = (kb2 == qblk);
#pragma unroll
    for (int nt = 0; nt < 4; ++nt) {
      v8f c;
#pragma unroll
      for (int j = 0; j < 8; ++j) c[j] = 0.0f;
#pragma unroll
      for (int ks = 0; ks < 4; ++ks) {
        AFrag kf;
        const bf16* kp = &Ks[(nt * 16 + l16) * KS_LD + ks * 32 + lh * 8];
        kf.h[0] = *reinterpret_cast<const v8bf*>(kp);
        kf.h[1] = *reinterpret_cast<const v8bf*>(kp + 16);
        c = wmma_bf16(qf[ks].v, kf.v, c);
      }
#pragma unroll
      for (int r = 0; r < 8; ++r) {
        float v = c[r] * scale;
        // C layout: key col = nt*16+l16, query row = wid*16 + lh*8 + r
        if (diag && (nt * 16 + l16 > wid * 16 + lh * 8 + r)) v = -1e30f;
        sc[nt][r] = v;
      }
    }

    // online softmax (row reductions across the 16-lane half via xor shuffles)
    float alpha[8];
#pragma unroll
    for (int r = 0; r < 8; ++r) {
      float rmax = fmaxf(fmaxf(sc[0][r], sc[1][r]), fmaxf(sc[2][r], sc[3][r]));
      rmax = fmaxf(rmax, __shfl_xor(rmax, 1));
      rmax = fmaxf(rmax, __shfl_xor(rmax, 2));
      rmax = fmaxf(rmax, __shfl_xor(rmax, 4));
      rmax = fmaxf(rmax, __shfl_xor(rmax, 8));
      const float mnew = fmaxf(m[r], rmax);
      alpha[r] = __expf(m[r] - mnew);
      m[r] = mnew;
    }
#pragma unroll
    for (int nt = 0; nt < 4; ++nt)
#pragma unroll
      for (int r = 0; r < 8; ++r) sc[nt][r] = __expf(sc[nt][r] - m[r]);
#pragma unroll
    for (int r = 0; r < 8; ++r) {
      float rs = sc[0][r] + sc[1][r] + sc[2][r] + sc[3][r];
      rs += __shfl_xor(rs, 1);
      rs += __shfl_xor(rs, 2);
      rs += __shfl_xor(rs, 4);
      rs += __shfl_xor(rs, 8);
      ssum[r] = ssum[r] * alpha[r] + rs;
    }

    // C-layout -> A-layout reshape of P through per-wave LDS tile (bf16)
    bf16* pws = &Ps[wid * 16 * PS_LD];
#pragma unroll
    for (int nt = 0; nt < 4; ++nt)
#pragma unroll
      for (int r = 0; r < 8; ++r)
        pws[(lh * 8 + r) * PS_LD + nt * 16 + l16] = (bf16)sc[nt][r];

    // rescale running output
#pragma unroll
    for (int n = 0; n < 8; ++n)
#pragma unroll
      for (int r = 0; r < 8; ++r) o[n][r] *= alpha[r];

    // O += P (16x64) * V (64x128): 2 k-steps x 8 dk n-tiles
#pragma unroll
    for (int ks2 = 0; ks2 < 2; ++ks2) {
      AFrag pf;
      const bf16* pp = &pws[l16 * PS_LD + ks2 * 32 + lh * 8];
      pf.h[0] = *reinterpret_cast<const v8bf*>(pp);
      pf.h[1] = *reinterpret_cast<const v8bf*>(pp + 16);
#pragma unroll
      for (int n = 0; n < 8; ++n) {
        AFrag vf;
        const bf16* vp = &Vt[(n * 16 + l16) * VT_LD + ks2 * 32 + lh * 8];
        vf.h[0] = *reinterpret_cast<const v8bf*>(vp);
        vf.h[1] = *reinterpret_cast<const v8bf*>(vp + 16);
        o[n] = wmma_bf16(pf.v, vf.v, o[n]);
      }
    }
  }

  // normalize and write bf16 output rows [B*S, 2048], head-concat layout
#pragma unroll
  for (int r = 0; r < 8; ++r) {
    const float inv = 1.0f / ssum[r];
    const size_t row = (size_t)(b * S + qblk * 64 + wid * 16 + lh * 8 + r);
#pragma unroll
    for (int n = 0; n < 8; ++n)
      attout[row * 2048 + h * 128 + n * 16 + l16] = (bf16)(o[n][r] * inv);
  }
}

// ---------------------------------------------------------------------------
extern "C" void kernel_launch(void* const* d_in, const int* in_sizes, int n_in,
                              void* d_out, int out_size, void* d_ws,
                              size_t ws_size, hipStream_t stream) {
  (void)in_sizes; (void)n_in; (void)out_size; (void)ws_size;
  const float* x    = (const float*)d_in[0];  // [2,2048,2048]
  const float* Wqkv = (const float*)d_in[1];  // [6144,2048]
  const float* Wo   = (const float*)d_in[2];  // [2048,2048]
  float* out = (float*)d_out;                 // [2,2048,2048] f32

  const size_t nX   = (size_t)4096 * 2048;
  const size_t nWq  = (size_t)6144 * 2048;
  const size_t nWo  = (size_t)2048 * 2048;
  const size_t nQKV = (size_t)4096 * 6144;

  bf16* xb   = (bf16*)d_ws;
  bf16* wqb  = xb + nX;
  bf16* wob  = wqb + nWq;
  bf16* qkvb = wob + nWo;
  bf16* attb = qkvb + nQKV;  // total ws use ~112 MB

  cast_kernel<<<(unsigned)((nX / 4 + 255) / 256), 256, 0, stream>>>(x, xb, (long)nX);
  cast_kernel<<<(unsigned)((nWq / 4 + 255) / 256), 256, 0, stream>>>(Wqkv, wqb, (long)nWq);
  cast_kernel<<<(unsigned)((nWo / 4 + 255) / 256), 256, 0, stream>>>(Wo, wob, (long)nWo);

  // QKV projection: [4096,2048] x [6144,2048]^T -> bf16 [4096,6144]
  gemm_nt<false><<<dim3(6144 / 64, 4096 / 128), 256, 0, stream>>>(
      xb, wqb, qkvb, 6144, 2048);

  // causal flash attention per (b,h,qblock)
  flash_attn<<<dim3(32, 16, 2), 128, 0, stream>>>(qkvb, attb);

  // output projection: [4096,2048] x [2048,2048]^T -> f32 d_out
  gemm_nt<true><<<dim3(2048 / 64, 4096 / 128), 256, 0, stream>>>(
      attb, wob, out, 2048, 2048);
}
